// GIN_8108898255053
// MI455X (gfx1250) — matrix operand order repaired and checked by reading, not compiled, not run on hardware
//
#include <hip/hip_runtime.h>

// ---------------------------------------------------------------------------
// GIN (2 layers, D=128) for MI455X / gfx1250, wave32 + WMMA bf16.
//   x      = h + segment_sum(h[src], dst)        (scatter kernel, f32 atomics)
//   h'     = relu(x @ Wa + ba) @ Wb + bb         (fused WMMA MLP kernel)
// ---------------------------------------------------------------------------

#define GIN_D  128   // feature dim
#define GIN_MT 128   // rows (nodes) per block in the MLP kernel

typedef __attribute__((ext_vector_type(16))) __bf16 v16bf;
typedef __attribute__((ext_vector_type(8)))  float  v8f;

// ---------------------------------------------------------------- copy x = h
__global__ void gin_copy_kernel(const float4* __restrict__ in,
                                float4* __restrict__ out, int n4) {
  int i = blockIdx.x * blockDim.x + threadIdx.x;
  int stride = gridDim.x * blockDim.x;
  for (; i < n4; i += stride) out[i] = in[i];
}

// ------------------------------------------------- agg[dst] += h[src] (edges)
// One wave per edge; lane moves 4 floats (32 lanes * 4 = 128 = D).
__global__ void gin_scatter_kernel(const float* __restrict__ h,
                                   const int* __restrict__ src,
                                   const int* __restrict__ dst,
                                   float* __restrict__ agg, int n_edges) {
  const int e = blockIdx.x * 8 + (threadIdx.x >> 5);
  if (e >= n_edges) return;
  const int lane = threadIdx.x & 31;
  const int s = src[e];
  const int d = dst[e];
  const float4 v = ((const float4*)(h + (size_t)s * GIN_D))[lane];
  float* ap = agg + (size_t)d * GIN_D + lane * 4;
  atomicAdd(ap + 0, v.x);
  atomicAdd(ap + 1, v.y);
  atomicAdd(ap + 2, v.z);
  atomicAdd(ap + 3, v.w);
}

// ------------------------------------------- fused MLP: relu(xWa+ba)Wb+bb
// Block = 256 threads = 8 waves; wave owns a 16x128 row slab.
// Per 16x16 output tile: 4x v_wmma_f32_16x16x32_bf16 (K = 128).
__global__ __launch_bounds__(256) void gin_mlp_kernel(
    const float* __restrict__ x,
    const float* __restrict__ Wa, const float* __restrict__ ba,
    const float* __restrict__ Wb, const float* __restrict__ bb,
    float* __restrict__ out, int n_rows, int relu_out) {
  // sW holds Wa^T (bf16) for GEMM1, then is reloaded with Wb^T for GEMM2.
  __shared__ __align__(32) __bf16 sW[GIN_D * GIN_D];   // 32 KB
  __shared__ __align__(32) __bf16 sH[GIN_MT * GIN_D];  // 32 KB hidden acts

  const int tid = threadIdx.x;

  // Stage Wa^T: sW[n*D + k] = bf16(Wa[k*D + n]); global reads coalesced.
  for (int i = tid; i < GIN_D * GIN_D; i += 256) {
    const int k = i >> 7, n = i & 127;
    sW[n * GIN_D + k] = (__bf16)Wa[i];
  }
  __syncthreads();

  const int wave  = tid >> 5;
  const int lane  = tid & 31;
  const int lrow  = lane & 15;   // row within wave slab (A) / column (B,C)
  const int khalf = lane >> 4;   // ISA half-lane K split
  const int row0  = blockIdx.x * GIN_MT + wave * 16;

  // --- A fragments for GEMM1 from global x (clamped tail rows) -------------
  int arow = row0 + lrow;
  if (arow >= n_rows) arow = n_rows - 1;
  v16bf afrag[4];
#pragma unroll
  for (int kk = 0; kk < 4; ++kk) {
    // lane holds K = 16*khalf+0..15 within this K32 block
    const float* xp = x + (size_t)arow * GIN_D + kk * 32 + khalf * 16;
#pragma unroll
    for (int e = 0; e < 8; ++e) {
      afrag[kk][e]     = (__bf16)xp[e];
      afrag[kk][e + 8] = (__bf16)xp[e + 8];
    }
  }

  // --- GEMM1: acc[nt] = x @ Wa over 8 N-tiles ------------------------------
  v8f acc[8];
#pragma unroll
  for (int nt = 0; nt < 8; ++nt) {
    v8f c = {};
    const int n = nt * 16 + lrow;
#pragma unroll
    for (int kk = 0; kk < 4; ++kk) {
      // B layout: lane = column n, K = 16*khalf + 0..15 (contiguous 32B)
      v16bf b = *(const v16bf*)&sW[n * GIN_D + kk * 32 + khalf * 16];
      c = __builtin_amdgcn_wmma_f32_16x16x32_bf16(false, afrag[kk], false, b,
                                                  (short)0, c, false, false);
    }
    acc[nt] = c;
  }
  __syncthreads();  // everyone done reading Wa^T from sW

  // Stage Wb^T into the same LDS buffer.
  for (int i = tid; i < GIN_D * GIN_D; i += 256) {
    const int k = i >> 7, n = i & 127;
    sW[n * GIN_D + k] = (__bf16)Wb[i];
  }

  // hidden = relu(acc + ba) -> sH (bf16), using the f32 C/D lane layout:
  // vgpr r, lane l  ->  row r + 8*(l>>4), col l&15.
#pragma unroll
  for (int nt = 0; nt < 8; ++nt) {
    const int n = nt * 16 + lrow;
    const float bias = ba[n];
#pragma unroll
    for (int r = 0; r < 8; ++r) {
      const int m = r + khalf * 8;
      float v = acc[nt][r] + bias;
      v = v > 0.0f ? v : 0.0f;
      sH[(wave * 16 + m) * GIN_D + n] = (__bf16)v;
    }
  }
  __syncthreads();

  // --- A fragments for GEMM2 from sH ---------------------------------------
  v16bf afrag2[4];
#pragma unroll
  for (int kk = 0; kk < 4; ++kk) {
    const __bf16* hp = &sH[(wave * 16 + lrow) * GIN_D + kk * 32 + khalf * 16];
#pragma unroll
    for (int e = 0; e < 8; ++e) {
      afrag2[kk][e]     = hp[e];
      afrag2[kk][e + 8] = hp[e + 8];
    }
  }

  // --- GEMM2: out = hidden @ Wb + bb (+ optional relu), f32 stores ---------
#pragma unroll
  for (int nt = 0; nt < 8; ++nt) {
    v8f c = {};
    const int n = nt * 16 + lrow;
#pragma unroll
    for (int kk = 0; kk < 4; ++kk) {
      v16bf b = *(const v16bf*)&sW[n * GIN_D + kk * 32 + khalf * 16];
      c = __builtin_amdgcn_wmma_f32_16x16x32_bf16(false, afrag2[kk], false, b,
                                                  (short)0, c, false, false);
    }
    const float bias = bb[n];
#pragma unroll
    for (int r = 0; r < 8; ++r) {
      const int m = r + khalf * 8;
      const int grow = row0 + m;
      float v = c[r] + bias;
      if (relu_out) v = v > 0.0f ? v : 0.0f;
      if (grow < n_rows) out[(size_t)grow * GIN_D + n] = v;
    }
  }
}

// ---------------------------------------------------------------------------
extern "C" void kernel_launch(void* const* d_in, const int* in_sizes, int n_in,
                              void* d_out, int out_size, void* d_ws, size_t ws_size,
                              hipStream_t stream) {
  const float* inputs = (const float*)d_in[0];
  const int*   src    = (const int*)d_in[1];
  const int*   dst    = (const int*)d_in[2];
  const float* W1a    = (const float*)d_in[3];
  const float* b1a    = (const float*)d_in[4];
  const float* W1b    = (const float*)d_in[5];
  const float* b1b    = (const float*)d_in[6];
  const float* W2a    = (const float*)d_in[7];
  const float* b2a    = (const float*)d_in[8];
  const float* W2b    = (const float*)d_in[9];
  const float* b2b    = (const float*)d_in[10];

  const int    n_nodes = in_sizes[0] / GIN_D;
  const int    n_edges = in_sizes[1];
  const size_t nd      = (size_t)n_nodes * GIN_D;

  float* xbuf = (float*)d_ws;        // n_nodes * D  (x = h + agg)
  float* hmid = (float*)d_ws + nd;   // n_nodes * D  (layer-1 output)

  const int n4 = (int)(nd / 4);
  dim3 cgrd(1024), cblk(256);
  dim3 sgrd((n_edges + 7) / 8), sblk(256);
  dim3 mgrd((n_nodes + GIN_MT - 1) / GIN_MT), mblk(256);

  // ---- layer 1: x1 = inputs + segsum(inputs[src], dst); h = relu(MLP(x1))
  gin_copy_kernel<<<cgrd, cblk, 0, stream>>>((const float4*)inputs, (float4*)xbuf, n4);
  gin_scatter_kernel<<<sgrd, sblk, 0, stream>>>(inputs, src, dst, xbuf, n_edges);
  gin_mlp_kernel<<<mgrd, mblk, 0, stream>>>(xbuf, W1a, b1a, W1b, b1b, hmid, n_nodes, 1);

  // ---- layer 2: x2 = h + segsum(h[src], dst); out = MLP(x2)
  gin_copy_kernel<<<cgrd, cblk, 0, stream>>>((const float4*)hmid, (float4*)xbuf, n4);
  gin_scatter_kernel<<<sgrd, sblk, 0, stream>>>(hmid, src, dst, xbuf, n_edges);
  gin_mlp_kernel<<<mgrd, mblk, 0, stream>>>(xbuf, W2a, b2a, W2b, b2b, (float*)d_out, n_nodes, 0);
}
